// MultiHeadAttention_8117488190177
// MI455X (gfx1250) — matrix operand order repaired and checked by reading, not compiled, not run on hardware
//
#include <hip/hip_runtime.h>
#include <hip/hip_bf16.h>

typedef __attribute__((ext_vector_type(16))) _Float16 v16h;
typedef __attribute__((ext_vector_type(2)))  _Float16 v2h;
typedef __attribute__((ext_vector_type(8)))  float    v8f;
typedef __attribute__((ext_vector_type(4)))  unsigned int u32x4;
typedef __attribute__((ext_vector_type(8)))  int i32x8;
typedef __attribute__((ext_vector_type(4)))  int i32x4;

#define BATCH  4
#define SEQ    2048
#define DMODEL 256
#define NHEAD  4
#define HDIM   64
// (1/sqrt(HDIM)) * log2(e): fold softmax scale + base-2 exp into Q
#define QSCALE 0.18033688011112042f

// A-matrix (16x32 f16) per-lane K offset for VGPR v (pair of halves)
__device__ __forceinline__ int a_koff(int v, int kh) {
  return ((v & 3) * 2) + ((v >= 4) ? 16 : 0) + kh * 8;
}
// B-matrix (32x16 f16) per-lane K offset for VGPR v (pair of halves)
__device__ __forceinline__ int b_koff(int v, int kh) {
  return ((v & 3) * 2) + ((v >= 4) ? 16 : 0) + kh * 8;
}

// ---------------------------------------------------------------------------
// Tensor DMA descriptor packing (CDNA5 ISA).
// group0: count=1 | lds_addr | global_addr[56:0] | type=2
// group1: data_size=8B, 2D tile (tile_dim2=0), huge tensor dims (no OOB).
// ---------------------------------------------------------------------------
__device__ __forceinline__ u32x4 tdm_group0(unsigned lds_addr,
                                            unsigned long long gaddr) {
  u32x4 g;
  g[0] = 1u;                                           // count=1, user mode
  g[1] = lds_addr;                                     // bits 63:32
  g[2] = (unsigned)(gaddr & 0xffffffffull);            // bits 95:64
  g[3] = (unsigned)((gaddr >> 32) & 0x01ffffffull)     // addr[56:32] -> 120:96
       | (2u << 30);                                   // type=2 ("image")
  return g;
}

__device__ __forceinline__ i32x8 tdm_group1(unsigned tile_d0, unsigned tile_d1,
                                            unsigned long long stride0_units) {
  const unsigned td = 0x40000000u;                     // huge tensor dims
  i32x8 g;
  g[0] = (int)(3u << 16);                              // data_size = 8 bytes
  g[1] = (int)((td & 0xffffu) << 16);                  // tensor_dim0[15:0]
  g[2] = (int)(((td >> 16) & 0xffffu) |                // tensor_dim0[31:16]
               ((td & 0xffffu) << 16));                // tensor_dim1[15:0]
  g[3] = (int)(((td >> 16) & 0xffffu) |                // tensor_dim1[31:16]
               ((tile_d0 & 0xffffu) << 16));           // tile_dim0
  g[4] = (int)(tile_d1 & 0xffffu);                     // tile_dim1; tile_dim2=0
  g[5] = (int)(stride0_units & 0xffffffffull);         // tensor_dim0_stride lo
  g[6] = (int)((stride0_units >> 32) & 0xffffull);     // stride hi; dim1_stride=0
  g[7] = 0;
  return g;
}

__device__ __forceinline__ void tdm_load(u32x4 g0, i32x8 g1) {
  const i32x4 z4 = {0, 0, 0, 0};
#if __has_include(<hip/amd_detail/amd_gfx1250_TDM.h>)
  const i32x8 z8 = {0, 0, 0, 0, 0, 0, 0, 0};
  __builtin_amdgcn_tensor_load_to_lds(g0, g1, z4, z4, z8, 0);   // clang-23 arity
#else
  __builtin_amdgcn_tensor_load_to_lds(g0, g1, z4, z4, 0);       // ROCm 7.2 arity
#endif
}

// ---------------------------------------------------------------------------
// Kernel 1: fused Q/K/V projection.  x[8192x256]f32 @ W^T -> f16 tensors.
// Q scaled by QSCALE, layout [bh][s][d]; K layout [bh][s][d]; V stored
// transposed [bh][d][s] so attention P@V B-fragments are contiguous.
// ---------------------------------------------------------------------------
__global__ __launch_bounds__(256) void qkv_proj_kernel(
    const float* __restrict__ x,  const float* __restrict__ Wq,
    const float* __restrict__ Wk, const float* __restrict__ Wv,
    _Float16* __restrict__ Q, _Float16* __restrict__ K, _Float16* __restrict__ VT)
{
  const int lane = threadIdx.x & 31;
  const int wave = threadIdx.x >> 5;
  const int mrow = lane & 15;
  const int kh   = lane >> 4;
  const int m0   = blockIdx.x * 16;

  // Cache full A row-slab (16 tokens x 256) as 8 k-step fragments (64 VGPRs)
  v16h a[8];
  const float* xr = x + (size_t)(m0 + mrow) * DMODEL;
#pragma unroll
  for (int ks = 0; ks < 8; ++ks)
#pragma unroll
    for (int v = 0; v < 8; ++v) {
      float2 f = *(const float2*)(xr + ks * 32 + a_koff(v, kh));
      a[ks][2 * v]     = (_Float16)f.x;
      a[ks][2 * v + 1] = (_Float16)f.y;
    }

  // 48 jobs = {Q,K,V} x 16 n-tiles, 6 per wave
  for (int job = wave * 6; job < wave * 6 + 6; ++job) {
    const int mat = job >> 4;       // 0=Q 1=K 2=V
    const int nt  = job & 15;
    const float* W = (mat == 0) ? Wq : (mat == 1) ? Wk : Wv;
    const int ncol = nt * 16 + mrow;
    const float* wr = W + (size_t)ncol * DMODEL;
    v8f c = {};
#pragma unroll
    for (int ks = 0; ks < 8; ++ks) {
      v16h b;
#pragma unroll
      for (int v = 0; v < 8; ++v) {
        float2 f = *(const float2*)(wr + ks * 32 + b_koff(v, kh));
        b[2 * v]     = (_Float16)f.x;
        b[2 * v + 1] = (_Float16)f.y;
      }
      c = __builtin_amdgcn_wmma_f32_16x16x32_f16(false, a[ks], false, b,
                                                 (short)0, c, false, false);
    }
    const int h = ncol >> 6, d = ncol & 63;
#pragma unroll
    for (int r = 0; r < 8; ++r) {
      const int t  = m0 + r + kh * 8;
      const int bb = t >> 11;
      const int s  = t & (SEQ - 1);
      const size_t bh = (size_t)(bb * NHEAD + h);
      if (mat == 0)      Q [(bh * SEQ + s) * HDIM + d] = (_Float16)(c[r] * QSCALE);
      else if (mat == 1) K [(bh * SEQ + s) * HDIM + d] = (_Float16)c[r];
      else               VT[(bh * HDIM + d) * SEQ + s] = (_Float16)c[r];
    }
  }
}

// ---------------------------------------------------------------------------
// Kernel 2: flash attention with CDNA5 Tensor Data Mover staging.
// 4 waves share one (b,h) 64-row Q block.  Each kv step, wave 0 issues a TDM
// descriptor for the K tile (32x64 f16, one contiguous 4 KB row) and wave 1
// for the V^T tile (64 rows x 64 B, pitch 4 KB); double-buffered in LDS and
// drained with s_wait_tensorcnt so the next tile's DMA overlaps compute.
// ---------------------------------------------------------------------------
// LDS layout (halves):  kbuf[2][32][64] | vbuf[2][64][32] | pbuf[4][16][40]
#define KBUF_OFF 0
#define VBUF_OFF 4096
#define PBUF_OFF 8192
#define SMEM_HALVES (PBUF_OFF + 4 * 16 * 40)

__global__ __launch_bounds__(128) void attn_kernel(
    const _Float16* __restrict__ Q, const _Float16* __restrict__ K,
    const _Float16* __restrict__ VT, const int* __restrict__ mask,
    _Float16* __restrict__ O)
{
  __shared__ _Float16 smem[SMEM_HALVES];
  const int tid  = threadIdx.x;
  const int lane = tid & 31;
  const int wave = tid >> 5;
  const int mrow = lane & 15;
  const int kh   = lane >> 4;
  const int qb   = blockIdx.x;
  const int bh   = qb >> 5;                  // 32 blocks of 64 rows per head
  const int q0   = (qb & 31) * 64 + wave * 16;
  const int b    = bh >> 2;
  const int h    = bh & 3;

  const unsigned k_lds0 = (unsigned)(uintptr_t)&smem[KBUF_OFF];
  const unsigned v_lds0 = (unsigned)(uintptr_t)&smem[VBUF_OFF];
  const unsigned long long kg =
      (unsigned long long)(uintptr_t)(K + (size_t)bh * SEQ * HDIM);
  const unsigned long long vg =
      (unsigned long long)(uintptr_t)(VT + (size_t)bh * HDIM * SEQ);

  // Q fragments: 16x64 = two 16x32 A tiles (pre-scaled f16, contiguous pairs)
  v16h aq[2];
  const _Float16* qr = Q + ((size_t)bh * SEQ + q0 + mrow) * HDIM;
#pragma unroll
  for (int ks = 0; ks < 2; ++ks)
#pragma unroll
    for (int v = 0; v < 8; ++v) {
      v2h e = *(const v2h*)(qr + ks * 32 + a_koff(v, kh));
      aq[ks][2 * v] = e[0]; aq[ks][2 * v + 1] = e[1];
    }

  float rowm[8], rowl[8];
  v8f acc[4];
  const v8f vzero = {};
#pragma unroll
  for (int r = 0; r < 8; ++r) { rowm[r] = -1e30f; rowl[r] = 0.0f; }
#pragma unroll
  for (int nt = 0; nt < 4; ++nt) acc[nt] = vzero;

  const int* mk = mask + (size_t)b * SEQ;

  // One TDM descriptor per tile: wave 0 -> K tile, wave 1 -> V^T tile.
  auto issue_tile = [&](int buf, int j0) {
    if (wave == 0) {
      // K tile: contiguous 4 KB = 512 x 8B, single row
      tdm_load(tdm_group0(k_lds0 + (unsigned)buf * 4096u,
                          kg + (unsigned long long)(j0 * HDIM * 2)),
               tdm_group1(512u, 1u, 512ull));
    } else if (wave == 1) {
      // V^T tile: 64 rows x 8 x 8B, global row pitch = SEQ*2 = 4096 B = 512 units
      tdm_load(tdm_group0(v_lds0 + (unsigned)buf * 4096u,
                          vg + (unsigned long long)(j0 * 2)),
               tdm_group1(8u, 64u, 512ull));
    }
  };

  issue_tile(0, 0);

  const int nsteps = SEQ / 32;
  for (int step = 0; step < nsteps; ++step) {
    const int j0  = step * 32;
    const int buf = step & 1;
    if (step + 1 < nsteps) {
      issue_tile(buf ^ 1, j0 + 32);
      __builtin_amdgcn_s_wait_tensorcnt(1);  // this tile done, prefetch in flight
    } else {
      __builtin_amdgcn_s_wait_tensorcnt(0);
    }
    __syncthreads();   // both DMA'd tiles visible to all waves

    // ---- scores 16x32 : two 16x16 WMMA C tiles, K-dim 64 split in 2 ----
    v8f sc[2];
#pragma unroll
    for (int t = 0; t < 2; ++t) {
      const _Float16* kr = smem + KBUF_OFF + buf * 2048 + (t * 16 + mrow) * 64;
      v8f cz = vzero;
#pragma unroll
      for (int ks = 0; ks < 2; ++ks) {
        v16h bk;
#pragma unroll
        for (int v = 0; v < 8; ++v) {
          v2h e = *(const v2h*)(kr + ks * 32 + b_koff(v, kh));
          bk[2 * v] = e[0]; bk[2 * v + 1] = e[1];
        }
        cz = __builtin_amdgcn_wmma_f32_16x16x32_f16(false, aq[ks], false, bk,
                                                    (short)0, cz, false, false);
      }
      if (mk[j0 + t * 16 + mrow] == 0) {
#pragma unroll
        for (int r = 0; r < 8; ++r) cz[r] = -1e30f;
      }
      sc[t] = cz;
    }

    // ---- online softmax (base 2); row = r + kh*8, cols across 16 lanes ----
    float alpha[8];
#pragma unroll
    for (int r = 0; r < 8; ++r) {
      float mx = fmaxf(sc[0][r], sc[1][r]);
#pragma unroll
      for (int off = 8; off; off >>= 1) mx = fmaxf(mx, __shfl_xor(mx, off, 32));
      const float mnew = fmaxf(rowm[r], mx);
      const float al = exp2f(rowm[r] - mnew);
      const float p0 = exp2f(sc[0][r] - mnew);
      const float p1 = exp2f(sc[1][r] - mnew);
      sc[0][r] = p0; sc[1][r] = p1;
      float ps = p0 + p1;
#pragma unroll
      for (int off = 8; off; off >>= 1) ps += __shfl_xor(ps, off, 32);
      rowl[r] = rowl[r] * al + ps;
      rowm[r] = mnew;
      alpha[r] = al;
    }
#pragma unroll
    for (int nt = 0; nt < 4; ++nt)
#pragma unroll
      for (int r = 0; r < 8; ++r) acc[nt][r] *= alpha[r];

    // ---- re-layout P: C-layout -> A-layout through per-wave LDS tile ----
    _Float16* pb = smem + PBUF_OFF + wave * 640;           // [16][40]
#pragma unroll
    for (int t = 0; t < 2; ++t)
#pragma unroll
      for (int r = 0; r < 8; ++r)
        pb[(r + kh * 8) * 40 + t * 16 + mrow] = (_Float16)sc[t][r];
    asm volatile("s_wait_dscnt 0" ::: "memory");
    v16h ap;
#pragma unroll
    for (int v = 0; v < 8; ++v) {
      v2h e = *(const v2h*)(pb + mrow * 40 + a_koff(v, kh));
      ap[2 * v] = e[0]; ap[2 * v + 1] = e[1];
    }

    // ---- acc += P @ V ; V^T tile rows contiguous in kv index ----
#pragma unroll
    for (int nt = 0; nt < 4; ++nt) {
      const _Float16* vr = smem + VBUF_OFF + buf * 2048 + (nt * 16 + mrow) * 32;
      v16h bv;
#pragma unroll
      for (int v = 0; v < 8; ++v) {
        v2h e = *(const v2h*)(vr + b_koff(v, kh));
        bv[2 * v] = e[0]; bv[2 * v + 1] = e[1];
      }
      acc[nt] = __builtin_amdgcn_wmma_f32_16x16x32_f16(false, ap, false, bv,
                                                       (short)0, acc[nt], false, false);
    }
    __syncthreads();   // everyone done with this buffer before it is refilled
  }

  // ---- normalize, store f16 [B][S][256] (heads interleaved per token) ----
#pragma unroll
  for (int nt = 0; nt < 4; ++nt) {
    const int f = h * HDIM + nt * 16 + mrow;
#pragma unroll
    for (int r = 0; r < 8; ++r) {
      const int s = q0 + r + kh * 8;
      O[((size_t)b * SEQ + s) * DMODEL + f] = (_Float16)(acc[nt][r] / rowl[r]);
    }
  }
}

// ---------------------------------------------------------------------------
// Kernel 3: output projection.  attn[8192x256]f16 @ Wo^T -> f32 d_out.
// ---------------------------------------------------------------------------
__global__ __launch_bounds__(256) void out_proj_kernel(
    const _Float16* __restrict__ A, const float* __restrict__ Wo,
    float* __restrict__ out)
{
  const int lane = threadIdx.x & 31;
  const int wave = threadIdx.x >> 5;
  const int mrow = lane & 15;
  const int kh   = lane >> 4;
  const int m0   = blockIdx.x * 16;

  v16h a[8];
  const _Float16* ar = A + (size_t)(m0 + mrow) * DMODEL;
#pragma unroll
  for (int ks = 0; ks < 8; ++ks)
#pragma unroll
    for (int v = 0; v < 8; ++v) {
      v2h e = *(const v2h*)(ar + ks * 32 + a_koff(v, kh));
      a[ks][2 * v] = e[0]; a[ks][2 * v + 1] = e[1];
    }

#pragma unroll
  for (int i = 0; i < 2; ++i) {
    const int nt   = wave * 2 + i;
    const int ncol = nt * 16 + mrow;
    const float* wr = Wo + (size_t)ncol * DMODEL;
    v8f c = {};
#pragma unroll
    for (int ks = 0; ks < 8; ++ks) {
      v16h b;
#pragma unroll
      for (int v = 0; v < 8; ++v) {
        float2 f = *(const float2*)(wr + ks * 32 + b_koff(v, kh));
        b[2 * v]     = (_Float16)f.x;
        b[2 * v + 1] = (_Float16)f.y;
      }
      c = __builtin_amdgcn_wmma_f32_16x16x32_f16(false, a[ks], false, b,
                                                 (short)0, c, false, false);
    }
#pragma unroll
    for (int r = 0; r < 8; ++r)
      out[((size_t)(m0 + r + kh * 8)) * DMODEL + ncol] = c[r];
  }
}

// ---------------------------------------------------------------------------
extern "C" void kernel_launch(void* const* d_in, const int* in_sizes, int n_in,
                              void* d_out, int out_size, void* d_ws, size_t ws_size,
                              hipStream_t stream) {
  (void)in_sizes; (void)n_in; (void)out_size; (void)ws_size;
  const float* x  = (const float*)d_in[0];
  const float* Wq = (const float*)d_in[1];
  const float* Wk = (const float*)d_in[2];
  const float* Wv = (const float*)d_in[3];
  const float* Wo = (const float*)d_in[4];
  const int* mask = (const int*)d_in[5];

  char* ws = (char*)d_ws;
  const size_t tbytes = (size_t)BATCH * SEQ * DMODEL * sizeof(_Float16); // 4 MB
  _Float16* Q  = (_Float16*)(ws);
  _Float16* K  = (_Float16*)(ws + tbytes);
  _Float16* VT = (_Float16*)(ws + 2 * tbytes);
  _Float16* AO = (_Float16*)(ws + 3 * tbytes);
  float* out = (float*)d_out;

  qkv_proj_kernel<<<dim3(BATCH * SEQ / 16), dim3(256), 0, stream>>>(x, Wq, Wk, Wv, Q, K, VT);
  attn_kernel<<<dim3(BATCH * NHEAD * SEQ / 64 / 4), dim3(128), 0, stream>>>(Q, K, VT, mask, AO);
  out_proj_kernel<<<dim3(BATCH * SEQ / 16), dim3(256), 0, stream>>>(AO, Wo, out);
}
